// GraphBaseBlock_68247030334290
// MI455X (gfx1250) — compile-verified, matrix operand order checked by reading, not run on hardware
//
#include <hip/hip_runtime.h>
#include <hip/hip_bf16.h>

#define N_NODES 100000
#define N_EDGES 300000
#define DIM     256
#define N_LAYERS 4
#define BN_EPS  1e-5f

#define ROW_TILES      10
#define ROWS_PER_BLOCK (ROW_TILES * 16)   // 160; 625 * 160 = 100000 exactly

typedef __attribute__((ext_vector_type(16))) __bf16 v16bf;
typedef __attribute__((ext_vector_type(8)))  float  v8f;

union Frag {
    uint4 u[2];
    v16bf bf;
};

__device__ __forceinline__ unsigned short f2bf(float f) {
    unsigned int u = __builtin_bit_cast(unsigned int, f);
    u += 0x7FFFu + ((u >> 16) & 1u);   // round-to-nearest-even
    return (unsigned short)(u >> 16);
}
__device__ __forceinline__ float bf2f(unsigned short h) {
    unsigned int u = ((unsigned int)h) << 16;
    return __builtin_bit_cast(float, u);
}

// ---------------------------------------------------------------------------
// Weights f32 row-major [k][n]  ->  bf16 column-major [n][k] per (layer, sel)
// ---------------------------------------------------------------------------
__global__ void __launch_bounds__(256) prep_weights_kernel(
    const float* __restrict__ W, const float* __restrict__ SW,
    unsigned short* __restrict__ wbf)
{
    size_t idx = (size_t)blockIdx.x * blockDim.x + threadIdx.x;
    if (idx >= (size_t)N_LAYERS * 2 * DIM * DIM) return;
    int which = (int)(idx / (DIM * DIM));      // layer*2 + sel
    int rem   = (int)(idx & (DIM * DIM - 1));
    int k = rem >> 8;
    int n = rem & (DIM - 1);
    int layer = which >> 1;
    const float* src = (which & 1) ? SW : W;
    float v = src[(size_t)layer * DIM * DIM + (size_t)k * DIM + n];
    wbf[(size_t)which * DIM * DIM + (size_t)n * DIM + k] = f2bf(v);
}

// ---------------------------------------------------------------------------
// f32 -> bf16 conversion (x -> h_bf at layer 0)
// ---------------------------------------------------------------------------
__global__ void __launch_bounds__(256) cvt_f32_bf16_kernel(
    const float* __restrict__ in, unsigned short* __restrict__ out, size_t n)
{
    size_t idx = (size_t)blockIdx.x * blockDim.x + threadIdx.x;
    if (idx < n) out[idx] = f2bf(in[idx]);
}

// ---------------------------------------------------------------------------
// Fused dual GEMM via V_WMMA_F32_16X16X32_BF16:
//   support  = h @ W   (stored bf16, row-major, for the SPMM gather)
//   selfterm = h @ SW  (stored f32, row-major; becomes the agg accumulator)
//
// Block = 512 threads (16 waves). Wave w owns column tile w for BOTH weight
// matrices; it preloads all 8 K-step B fragments of each into registers
// (128 VGPRs) and then sweeps 10 row tiles (160 rows). Inside the row loop
// only A fragments are loaded; the same 8KB A tile is shared by all 16
// waves, so it stays hot in WGP$.
// ---------------------------------------------------------------------------
__global__ void __launch_bounds__(512) gemm_dual_kernel(
    const unsigned short* __restrict__ hbf,
    const unsigned short* __restrict__ wcol,    // bf16 col-major W
    const unsigned short* __restrict__ swcol,   // bf16 col-major SW
    unsigned short* __restrict__ support_bf,
    float* __restrict__ selfterm)
{
    const int lane = threadIdx.x & 31;
    const int wave = threadIdx.x >> 5;   // 0..15 -> column tile
    const int half = lane >> 4;          // 0: K[0:8)+K[16:24)   1: K[8:16)+K[24:32)
    const int l16  = lane & 15;
    const int colBase = wave << 4;
    const int n = colBase + l16;

    // Register-resident B fragments, full K = 256, both matrices
    Frag bw[8], bs[8];
    {
        const unsigned short* bwcol = wcol  + (size_t)n * DIM;
        const unsigned short* bscol = swcol + (size_t)n * DIM;
#pragma unroll
        for (int kk = 0; kk < 8; ++kk) {
            const unsigned short* pw = bwcol + kk * 32 + half * 8;
            const unsigned short* ps = bscol + kk * 32 + half * 8;
            bw[kk].u[0] = *(const uint4*)(pw);  bw[kk].u[1] = *(const uint4*)(pw + 16);
            bs[kk].u[0] = *(const uint4*)(ps);  bs[kk].u[1] = *(const uint4*)(ps + 16);
        }
    }

    const int rowBlock = blockIdx.x * ROWS_PER_BLOCK;
    for (int rt = 0; rt < ROW_TILES; ++rt) {
        const int rowBase = rowBlock + (rt << 4);
        const unsigned short* arow = hbf + (size_t)(rowBase + l16) * DIM;
        if (rt + 1 < ROW_TILES)
            __builtin_prefetch(arow + 16 * DIM, 0, 0);   // next row tile

        v8f cw = {};
        v8f cs = {};
#pragma unroll
        for (int kk = 0; kk < 8; ++kk) {
            Frag a;
            const unsigned short* p = arow + kk * 32 + half * 8;
            a.u[0] = *(const uint4*)(p);
            a.u[1] = *(const uint4*)(p + 16);
            cw = __builtin_amdgcn_wmma_f32_16x16x32_bf16(
                     false, a.bf, false, bw[kk].bf, (short)0, cw, false, false);
            cs = __builtin_amdgcn_wmma_f32_16x16x32_bf16(
                     false, a.bf, false, bs[kk].bf, (short)0, cs, false, false);
        }

        // C/D layout: VGPR r -> M = r + 8*half, N = colBase + (lane & 15)
#pragma unroll
        for (int r = 0; r < 8; ++r) {
            const int m = rowBase + r + half * 8;
            support_bf[(size_t)m * DIM + n] = f2bf(cw[r]);
            selfterm [(size_t)m * DIM + n] = cs[r];
        }
    }
}

// ---------------------------------------------------------------------------
// SPMM: agg[rows[e], :] += vals[e] * support[cols[e], :]
// Random 512B row gathers hit L2 (support fits in 192MB L2); f32 atomics.
// ---------------------------------------------------------------------------
#define EDGES_PER_BLOCK 4
__global__ void __launch_bounds__(256) spmm_kernel(
    const int* __restrict__ rows, const int* __restrict__ cols,
    const float* __restrict__ vals, const unsigned short* __restrict__ sup,
    float* __restrict__ agg)
{
    const int f  = threadIdx.x;
    const int e0 = blockIdx.x * EDGES_PER_BLOCK;
#pragma unroll
    for (int i = 0; i < EDGES_PER_BLOCK; ++i) {
        const int e = e0 + i;
        if (e >= N_EDGES) return;
        const int   r = rows[e];
        const int   c = cols[e];
        const float v = vals[e];
        if (i + 1 < EDGES_PER_BLOCK && e + 1 < N_EDGES)
            __builtin_prefetch(&sup[(size_t)cols[e + 1] * DIM + f], 0, 0);
        const float s = bf2f(sup[(size_t)c * DIM + f]);
        atomicAdd(&agg[(size_t)r * DIM + f], v * s);
    }
}

// ---------------------------------------------------------------------------
// BatchNorm statistics: stats[f] = sum, stats[256+f] = sumsq (over nodes)
// ---------------------------------------------------------------------------
__global__ void __launch_bounds__(256) zero_stats_kernel(float* __restrict__ stats)
{
    int idx = blockIdx.x * blockDim.x + threadIdx.x;
    if (idx < 2 * DIM) stats[idx] = 0.0f;
}

__global__ void __launch_bounds__(256) bn_stats_kernel(
    const float* __restrict__ agg, float* __restrict__ stats)
{
    const int f = threadIdx.x;
    float s = 0.0f, s2 = 0.0f;
    for (int r = blockIdx.x; r < N_NODES; r += gridDim.x) {
        const float v = agg[(size_t)r * DIM + f];
        s  += v;
        s2 += v * v;
    }
    atomicAdd(&stats[f], s);
    atomicAdd(&stats[DIM + f], s2);
}

// ---------------------------------------------------------------------------
// BatchNorm apply + ReLU. Writes bf16 h for next layer, or f32 into the
// strided final output (d_out has row stride 512).
// ---------------------------------------------------------------------------
__global__ void __launch_bounds__(256) bn_apply_kernel(
    const float* __restrict__ agg, const float* __restrict__ stats,
    const float* __restrict__ gamma, const float* __restrict__ beta,
    unsigned short* __restrict__ hbf_out, float* __restrict__ f32_out,
    int out_stride)
{
    size_t idx = (size_t)blockIdx.x * blockDim.x + threadIdx.x;
    if (idx >= (size_t)N_NODES * DIM) return;
    const int    f = (int)(idx & (DIM - 1));
    const size_t n = idx >> 8;
    const float invN = 1.0f / (float)N_NODES;
    const float mean = stats[f] * invN;
    const float var  = stats[DIM + f] * invN - mean * mean;
    const float inv  = rsqrtf(var + BN_EPS);
    float v = gamma[f] * (agg[idx] - mean) * inv + beta[f];
    v = fmaxf(v, 0.0f);
    if (hbf_out) hbf_out[idx] = f2bf(v);
    if (f32_out) f32_out[n * (size_t)out_stride + f] = v;
}

// ---------------------------------------------------------------------------
// Final concat: d_out[:, 256:512] = x
// ---------------------------------------------------------------------------
__global__ void __launch_bounds__(256) concat_x_kernel(
    const float* __restrict__ x, float* __restrict__ out)
{
    size_t idx = (size_t)blockIdx.x * blockDim.x + threadIdx.x;
    if (idx >= (size_t)N_NODES * DIM) return;
    const int    f = (int)(idx & (DIM - 1));
    const size_t n = idx >> 8;
    out[n * (size_t)(2 * DIM) + DIM + f] = x[idx];
}

// ---------------------------------------------------------------------------
extern "C" void kernel_launch(void* const* d_in, const int* in_sizes, int n_in,
                              void* d_out, int out_size, void* d_ws, size_t ws_size,
                              hipStream_t stream)
{
    const float* x            = (const float*)d_in[0];
    const float* vals         = (const float*)d_in[1];
    const float* weights      = (const float*)d_in[2];
    const float* self_weights = (const float*)d_in[3];
    const float* gammas       = (const float*)d_in[4];
    const float* betas        = (const float*)d_in[5];
    const int*   rows         = (const int*)d_in[6];
    const int*   cols         = (const int*)d_in[7];
    float*       out          = (float*)d_out;

    // Workspace layout (all offsets 256B aligned)
    const size_t HB  = (size_t)N_NODES * DIM * sizeof(unsigned short); // 51,200,000
    const size_t AGG = (size_t)N_NODES * DIM * sizeof(float);          // 102,400,000
    char* ws = (char*)d_ws;
    unsigned short* h_bf  = (unsigned short*)(ws);
    unsigned short* sup   = (unsigned short*)(ws + HB);
    float*          agg   = (float*)(ws + 2 * HB);
    float*          stats = (float*)(ws + 2 * HB + AGG);
    unsigned short* wbf   = (unsigned short*)(ws + 2 * HB + AGG + 4096);

    const size_t ND = (size_t)N_NODES * DIM;

    // Pack all layer weights as bf16 column-major once per launch
    prep_weights_kernel<<<(N_LAYERS * 2 * DIM * DIM) / 256, 256, 0, stream>>>(
        weights, self_weights, wbf);

    // h0 = bf16(x)
    cvt_f32_bf16_kernel<<<(unsigned)(ND / 256), 256, 0, stream>>>(x, h_bf, ND);

    for (int l = 0; l < N_LAYERS; ++l) {
        const unsigned short* wcol  = wbf + (size_t)(2 * l)     * DIM * DIM;
        const unsigned short* swcol = wbf + (size_t)(2 * l + 1) * DIM * DIM;

        gemm_dual_kernel<<<N_NODES / ROWS_PER_BLOCK, 512, 0, stream>>>(
            h_bf, wcol, swcol, sup, agg);

        spmm_kernel<<<N_EDGES / EDGES_PER_BLOCK, 256, 0, stream>>>(
            rows, cols, vals, sup, agg);

        zero_stats_kernel<<<2, 256, 0, stream>>>(stats);
        bn_stats_kernel<<<512, 256, 0, stream>>>(agg, stats);

        if (l < N_LAYERS - 1) {
            bn_apply_kernel<<<(unsigned)(ND / 256), 256, 0, stream>>>(
                agg, stats, gammas + l * DIM, betas + l * DIM,
                h_bf, (float*)nullptr, 0);
        } else {
            bn_apply_kernel<<<(unsigned)(ND / 256), 256, 0, stream>>>(
                agg, stats, gammas + l * DIM, betas + l * DIM,
                (unsigned short*)nullptr, out, 2 * DIM);
        }
    }

    concat_x_kernel<<<(unsigned)(ND / 256), 256, 0, stream>>>(x, out);
}
